// L_loss_79328045957337
// MI455X (gfx1250) — matrix operand order repaired and checked by reading, not compiled, not run on hardware
//
#include <hip/hip_runtime.h>
#include <hip/hip_bf16.h>
#include <math.h>

typedef __attribute__((ext_vector_type(16))) _Float16 v16h;
typedef __attribute__((ext_vector_type(8)))  _Float16 v8h;
typedef __attribute__((ext_vector_type(8)))  float    v8f;
typedef __attribute__((ext_vector_type(8)))  int      v8i;

#define M_MARGIN_F 32.0f
#define LAMTA_F    1.0f
#define ROU_F      1.0f
#define EPSILUO_F  0.1f
#define NBLK_PAIR  1024
#define BITS       64

__device__ __forceinline__ float wave_sum(float v) {
#pragma unroll
  for (int off = 16; off > 0; off >>= 1) v += __shfl_xor(v, off, 32);
  return v;
}
__device__ __forceinline__ float wave_max(float v) {
#pragma unroll
  for (int off = 16; off > 0; off >>= 1) v = fmaxf(v, __shfl_xor(v, off, 32));
  return v;
}

// ---------------------------------------------------------------------------
// Kernel 1: convert Fi (f32, B x 64) -> Fh (f16) and sq[row] = ||f16 row||^2
// One wave per row (32 lanes x 2 elements).
// ---------------------------------------------------------------------------
__global__ __launch_bounds__(256) void prep_kernel(const float* __restrict__ Fi,
                                                   _Float16* __restrict__ Fh,
                                                   float* __restrict__ sq, int B) {
  const int wave = threadIdx.x >> 5;
  const int lane = threadIdx.x & 31;
  const int row  = blockIdx.x * 8 + wave;
  if (row >= B) return;
  const float2 v = ((const float2*)(Fi + (size_t)row * BITS))[lane];
  const _Float16 h0 = (_Float16)v.x;
  const _Float16 h1 = (_Float16)v.y;
  _Float16* dst = Fh + (size_t)row * BITS + lane * 2;
  dst[0] = h0;
  dst[1] = h1;
  const float f0 = (float)h0, f1 = (float)h1;
  float s = wave_sum(f0 * f0 + f1 * f1);
  if (lane == 0) sq[row] = s;
}

// ---------------------------------------------------------------------------
// Kernel 2: per-row CE (Yi and Ym) + BCE quantization term. One wave per row.
// ---------------------------------------------------------------------------
__device__ __forceinline__ float row_ce(const float* __restrict__ logits, int C,
                                        int label, int lane) {
  float mx = -INFINITY;
  for (int c = lane; c < C; c += 32) mx = fmaxf(mx, logits[c]);
  mx = wave_max(mx);
  float se = 0.0f;
  for (int c = lane; c < C; c += 32) se += expf(logits[c] - mx);
  se = wave_sum(se);
  return mx + logf(se) - logits[label];
}

__global__ __launch_bounds__(256) void ce_qua_kernel(const float* __restrict__ Ym,
                                                     const float* __restrict__ Yi,
                                                     const float* __restrict__ Fi,
                                                     const int* __restrict__ y,
                                                     float* __restrict__ partial,
                                                     int B, int C) {
  const int wave = threadIdx.x >> 5;
  const int lane = threadIdx.x & 31;
  const int row  = blockIdx.x * 8 + wave;
  __shared__ float smem[8];
  float total = 0.0f;
  if (row < B) {
    const int label = y[row];
    const float ce_sem = row_ce(Yi + (size_t)row * C, C, label, lane);
    const float ce_att = row_ce(Ym + (size_t)row * C, C, label, lane);
    float q = 0.0f;
#pragma unroll
    for (int k = 0; k < 2; ++k) {
      const float p  = Fi[(size_t)row * BITS + lane + 32 * k];
      const float lp = fmaxf(logf(p), -100.0f);
      const float l1 = fmaxf(log1pf(-p), -100.0f);
      q += -(p * lp + (1.0f - p) * l1);
    }
    q = wave_sum(q);
    total = (LAMTA_F * ce_sem + ROU_F * ce_att) / (float)B +
            EPSILUO_F * q / ((float)B * (float)BITS);
  }
  if (lane == 0) smem[wave] = total;
  __syncthreads();
  if (threadIdx.x == 0) {
    float s = 0.0f;
#pragma unroll
    for (int w = 0; w < 8; ++w) s += smem[w];
    partial[blockIdx.x] = s;
  }
}

// ---------------------------------------------------------------------------
// Kernel 3: pairwise contrastive term via WMMA f32_16x16x32_f16.
// Each wave owns a 16x64 strip (one i-tile x 4 j-tiles). Interior strips take
// a mask-free fast path; only strips touching the diagonal pay per-element
// triangular masking. 8-wide accumulators break the serial add chain.
// ---------------------------------------------------------------------------
template <bool EDGE>
__device__ __forceinline__ void tile_accum(const _Float16* __restrict__ Fh,
                                           const float* __restrict__ sq,
                                           const int* __restrict__ y,
                                           const v16h& A0, const v16h& A1,
                                           const v8f& sqi, const v8i& yi,
                                           int ti, int tj, int lm, int ka, int k0,
                                           v8f& acc8) {
  const int rowB = tj << 4;

  // B fragments: B[k][n] = Fh[rowB + n][k] -> two 8-half runs per lane (table order, matches A)
  const _Float16* pb = Fh + (size_t)(rowB + lm) * BITS + ka;
  const v16h B0 = __builtin_shufflevector(*(const v8h*)(pb), *(const v8h*)(pb + 16),
                                          0, 1, 2, 3, 4, 5, 6, 7, 8, 9, 10, 11, 12, 13, 14, 15);
  const v16h B1 = __builtin_shufflevector(*(const v8h*)(pb + 32), *(const v8h*)(pb + 48),
                                          0, 1, 2, 3, 4, 5, 6, 7, 8, 9, 10, 11, 12, 13, 14, 15);

  v8f c = {};
  c = __builtin_amdgcn_wmma_f32_16x16x32_f16(false, A0, false, B0, (short)0,
                                             c, false, false);
  c = __builtin_amdgcn_wmma_f32_16x16x32_f16(false, A1, false, B1, (short)0,
                                             c, false, false);

  const float sqj = sq[rowB + lm];
  const int   yj  = y[rowB + lm];

#pragma unroll
  for (int r = 0; r < 8; ++r) {
    float D = sqi[r] + sqj - 2.0f * c[r];
    D = fmaxf(D, 0.0f);
    const float p = (yi[r] == yj) ? D : fmaxf(M_MARGIN_F - D, 0.0f);
    if (EDGE) {
      const int m = r + ka;                       // row within tile for c[r]
      const bool keep = (tj > ti) || ((tj == ti) && (m < lm));  // strict i<j
      acc8[r] += keep ? p : 0.0f;
    } else {
      acc8[r] += p;
    }
  }
}

__global__ __launch_bounds__(256) void pair_kernel(const _Float16* __restrict__ Fh,
                                                   const float* __restrict__ sq,
                                                   const int* __restrict__ y,
                                                   float* __restrict__ partial,
                                                   int B) {
  const int lane       = threadIdx.x & 31;
  const int wave       = threadIdx.x >> 5;
  const int globalWave = blockIdx.x * 8 + wave;
  const int totalWaves = gridDim.x * 8;

  const int NT    = B >> 4;        // 512 tiles per dimension
  const int NS    = NT >> 2;       // 128 strips (4 tiles) per row
  const int TASKS = NT * NS;       // rectangular task space

  const int halfSel = lane >> 4;   // 0: lanes 0-15, 1: lanes 16-31
  const int lm      = lane & 15;   // m (A) or n (B) within tile
  const int ka      = halfSel * 8; // A-fragment K sub-offset
  const int k0      = halfSel * 16;// unused; B fragments also use ka

  v8f acc8 = {};

  for (int u = globalWave; u < TASKS; u += totalWaves) {
    const int ti     = u / NS;             // wave-uniform
    const int strip  = u - ti * NS;
    const int tjBase = strip << 2;
    if (tjBase + 3 < ti) continue;         // strip entirely below diagonal

    const int rowA = ti << 4;

    // A fragments for K=0..31 and K=32..63 (documented 16-bit A layout)
    const _Float16* pa = Fh + (size_t)(rowA + lm) * BITS + ka;
    const v8h a0lo = *(const v8h*)(pa);
    const v8h a0hi = *(const v8h*)(pa + 16);
    const v8h a1lo = *(const v8h*)(pa + 32);
    const v8h a1hi = *(const v8h*)(pa + 48);
    const v16h A0 = __builtin_shufflevector(a0lo, a0hi, 0, 1, 2, 3, 4, 5, 6, 7,
                                            8, 9, 10, 11, 12, 13, 14, 15);
    const v16h A1 = __builtin_shufflevector(a1lo, a1hi, 0, 1, 2, 3, 4, 5, 6, 7,
                                            8, 9, 10, 11, 12, 13, 14, 15);

    const v8f sqi = *(const v8f*)(sq + rowA + ka);
    const v8i yi  = *(const v8i*)(y + rowA + ka);

    if (tjBase > ti) {
      // fast path: all 4 tiles strictly above the diagonal, no masking
#pragma unroll
      for (int q = 0; q < 4; ++q)
        tile_accum<false>(Fh, sq, y, A0, A1, sqi, yi, ti, tjBase + q,
                          lm, ka, k0, acc8);
    } else {
      // edge path: strip contains / straddles the diagonal tile
#pragma unroll
      for (int q = 0; q < 4; ++q)
        tile_accum<true>(Fh, sq, y, A0, A1, sqi, yi, ti, tjBase + q,
                         lm, ka, k0, acc8);
    }
  }

  float acc = 0.0f;
#pragma unroll
  for (int r = 0; r < 8; ++r) acc += acc8[r];   // fixed order

  acc = wave_sum(acc);
  __shared__ float smem[8];
  if (lane == 0) smem[wave] = acc;
  __syncthreads();
  if (threadIdx.x == 0) {
    float s = 0.0f;
#pragma unroll
    for (int w = 0; w < 8; ++w) s += smem[w];
    partial[blockIdx.x] = s;
  }
}

// ---------------------------------------------------------------------------
// Kernel 4: deterministic fixed-order final reduction into d_out[0].
// ---------------------------------------------------------------------------
__global__ __launch_bounds__(256) void final_kernel(const float* __restrict__ pp,
                                                    int npp,
                                                    const float* __restrict__ pc,
                                                    int npc, float* __restrict__ out,
                                                    float pairScale) {
  __shared__ float smp[256];
  __shared__ float smc[256];
  float sp = 0.0f, sc = 0.0f;
  for (int i = threadIdx.x; i < npp; i += 256) sp += pp[i];
  for (int i = threadIdx.x; i < npc; i += 256) sc += pc[i];
  smp[threadIdx.x] = sp;
  smc[threadIdx.x] = sc;
  __syncthreads();
  for (int off = 128; off > 0; off >>= 1) {
    if (threadIdx.x < off) {
      smp[threadIdx.x] += smp[threadIdx.x + off];
      smc[threadIdx.x] += smc[threadIdx.x + off];
    }
    __syncthreads();
  }
  if (threadIdx.x == 0) out[0] = smp[0] * pairScale + smc[0];
}

// ---------------------------------------------------------------------------
extern "C" void kernel_launch(void* const* d_in, const int* in_sizes, int n_in,
                              void* d_out, int out_size, void* d_ws, size_t ws_size,
                              hipStream_t stream) {
  const float* Ym = (const float*)d_in[0];
  const float* Fi = (const float*)d_in[1];
  const float* Yi = (const float*)d_in[2];
  const int*   y  = (const int*)d_in[3];

  const int B = in_sizes[3];            // 8192
  const int C = in_sizes[0] / B;        // 100

  // workspace carve-up
  char* ws = (char*)d_ws;
  _Float16* Fh = (_Float16*)ws;                                   // B*64 f16 (1 MB)
  float* sq = (float*)(ws + (size_t)B * BITS * sizeof(_Float16)); // B f32
  float* pp = (float*)((char*)sq + (size_t)B * sizeof(float));    // NBLK_PAIR f32
  float* pc = pp + NBLK_PAIR;                                     // B/8 f32

  const int rowBlocks = B / 8;

  prep_kernel<<<rowBlocks, 256, 0, stream>>>(Fi, Fh, sq, B);
  ce_qua_kernel<<<rowBlocks, 256, 0, stream>>>(Ym, Yi, Fi, y, pc, B, C);
  pair_kernel<<<NBLK_PAIR, 256, 0, stream>>>(Fh, sq, y, pp, B);

  const float pairScale = 1.0f / (2.0f * (float)B * (float)(B - 1));
  final_kernel<<<1, 256, 0, stream>>>(pp, NBLK_PAIR, pc, rowBlocks,
                                      (float*)d_out, pairScale);
}
